// GATPairClassifier_16784732193027
// MI455X (gfx1250) — matrix operand order repaired and checked by reading, not compiled, not run on hardware
//
#include <hip/hip_runtime.h>

// ---------------------------------------------------------------------------
// GAT pair classifier on gfx1250 (MI455X). All heavy GEMMs use
// v_wmma_f32_16x16x32_bf16 (bf16 operands, f32 accumulate), register-blocked
// 16x64 per wave (4 accumulators -> 4 independent WMMAs per A fragment).
// Softmax / ELU / PReLU / BN stay in f32 VALU. Attention probabilities are
// generated on the fly inside the WMMA K-loop, exactly once per (i,j)
// (softmax over axis=1 uses monotonicity of leaky_relu:
//  colmax[j] = LR(max_i f1[i] + f2[j])).
// ---------------------------------------------------------------------------

#define ALPHA_SLOPE 0.2f
#define BN_EPS 1e-5f

typedef __attribute__((ext_vector_type(16))) __bf16       v16bf;
typedef __attribute__((ext_vector_type(8)))  float        v8f;
typedef __attribute__((ext_vector_type(8)))  unsigned int v8u;

static __device__ __forceinline__ unsigned short f2bf(float f) {
  unsigned int u = __float_as_uint(f);
  return (unsigned short)((u + 0x7FFFu + ((u >> 16) & 1u)) >> 16);  // RNE
}
static __device__ __forceinline__ float bf2f(unsigned short h) {
  return __uint_as_float(((unsigned int)h) << 16);
}
static __device__ __forceinline__ float lrelu(float x) { return x >= 0.f ? x : ALPHA_SLOPE * x; }
static __device__ __forceinline__ float eluf(float x)  { return x > 0.f ? x : (__expf(x) - 1.f); }

static __device__ __forceinline__ v8f wmma_bf16(const v8u& a, const v8u& b, const v8f& c) {
  return __builtin_amdgcn_wmma_f32_16x16x32_bf16(
      false, __builtin_bit_cast(v16bf, a),
      false, __builtin_bit_cast(v16bf, b),
      (short)0, c, false, false);
}

// ---------------------------------------------------------------------------
// K0: f32 -> bf16 raw conversion
// ---------------------------------------------------------------------------
__global__ void k_convert(const float* __restrict__ src, unsigned short* __restrict__ dst, int n) {
  int i = blockIdx.x * blockDim.x + threadIdx.x;
  if (i < n) dst[i] = f2bf(src[i]);
}

// ---------------------------------------------------------------------------
// 16x64 output stripe per wave: K-loop, one A fragment per step feeding four
// independent v_wmma_f32_16x16x32_bf16 (N-tiles 0..3). N is fixed at 64.
// A frag: lane -> M (=lane&15), vgpr v + half-wave -> K (ISA doc layout).
// B frag: lane -> N (=lane&15), vgpr v + half-wave -> K pair (k,k+1), same K order as A.
// ---------------------------------------------------------------------------
static __device__ __forceinline__ void gemm_stripe_n64(
    const unsigned short* __restrict__ A, int lda,
    const unsigned short* __restrict__ B,            // row-major [K, 64]
    int m0, int K, int lane,
    v8f& c0, v8f& c1, v8f& c2, v8f& c3) {
  const int half8 = (lane >> 4) * 8;
  const int mrow  = m0 + (lane & 15);
  const int kbl   = (lane & 15) + ((lane >> 4) << 4);
  for (int kk = 0; kk < K; kk += 32) {
    v8u au;
#pragma unroll
    for (int v = 0; v < 8; ++v) {
      int ka = kk + ((v < 4) ? 2 * v : 16 + 2 * (v - 4)) + half8;
      au[v] = *(const unsigned int*)(A + (size_t)mrow * lda + ka);
    }
    v8u bu0, bu1, bu2, bu3;
#pragma unroll
    for (int v = 0; v < 8; ++v) {
      int kb = kk + ((v < 4) ? 2 * v : 16 + 2 * (v - 4)) + half8;
      const unsigned short* r0 = B + (size_t)kb * 64 + (lane & 15);
      const unsigned short* r1 = r0 + 64;
      bu0[v] = (unsigned int)r0[0]  | ((unsigned int)r1[0]  << 16);
      bu1[v] = (unsigned int)r0[16] | ((unsigned int)r1[16] << 16);
      bu2[v] = (unsigned int)r0[32] | ((unsigned int)r1[32] << 16);
      bu3[v] = (unsigned int)r0[48] | ((unsigned int)r1[48] << 16);
    }
    c0 = wmma_bf16(au, bu0, c0);
    c1 = wmma_bf16(au, bu1, c1);
    c2 = wmma_bf16(au, bu2, c2);
    c3 = wmma_bf16(au, bu3, c3);
  }
}

// C layout: element r of v8f at (M = r + 8*(lane>=16), N = lane&15).
static __device__ __forceinline__ void store_stripe_bf16(
    unsigned short* __restrict__ out, int ldo, int m0, int n_off, int lane,
    const v8f& c0, const v8f& c1, const v8f& c2, const v8f& c3) {
  int n = lane & 15;
  int mbase = m0 + ((lane >> 4) << 3);
#pragma unroll
  for (int r = 0; r < 8; ++r) {
    unsigned short* row = out + (size_t)(mbase + r) * ldo + n_off + n;
    row[0]  = f2bf(c0[r]);
    row[16] = f2bf(c1[r]);
    row[32] = f2bf(c2[r]);
    row[48] = f2bf(c3[r]);
  }
}

// ---------------------------------------------------------------------------
// K1: Wh[h][b*512+i][64] = NF[b*512+i][0:256] @ W_heads[h]   (bf16 out)
// 8 heads * 1024 mtiles = 8192 waves -> 1024 blocks of 8 waves.
// ---------------------------------------------------------------------------
__global__ __launch_bounds__(256) void k_gemm1(const unsigned short* __restrict__ nfb,
                                               const unsigned short* __restrict__ whhb,
                                               unsigned short* __restrict__ whA) {
  int wave = (blockIdx.x * blockDim.x + threadIdx.x) >> 5;
  int lane = threadIdx.x & 31;
  int h    = wave >> 10;          // / 1024
  int mt   = wave & 1023;
  v8f c0 = {}, c1 = {}, c2 = {}, c3 = {};
  gemm_stripe_n64(nfb, 256, whhb + (size_t)h * 256 * 64, mt * 16, 256, lane, c0, c1, c2, c3);
  store_stripe_bf16(whA + (size_t)h * 16384 * 64, 64, mt * 16, 0, lane, c0, c1, c2, c3);
}

// ---------------------------------------------------------------------------
// K2: f1/f2 per (h, b*512+i): 64-wide dots with a_heads halves
// ---------------------------------------------------------------------------
__global__ void k_fdots(const unsigned short* __restrict__ whA, const float* __restrict__ a_heads,
                        float* __restrict__ f1, float* __restrict__ f2) {
  int idx = blockIdx.x * blockDim.x + threadIdx.x;   // 131072
  int h   = idx >> 14;
  const unsigned short* wr = whA + (size_t)idx * 64;
  const float* a = a_heads + h * 128;
  float s1 = 0.f, s2 = 0.f;
#pragma unroll 8
  for (int d = 0; d < 64; ++d) { float w = bf2f(wr[d]); s1 += w * a[d]; s2 += w * a[64 + d]; }
  f1[idx] = s1; f2[idx] = s2;
}

// ---------------------------------------------------------------------------
// K3a: f1max[hb] = max_i f1[hb][i]   (one block per (h,b))
// ---------------------------------------------------------------------------
__global__ __launch_bounds__(256) void k_f1max(const float* __restrict__ f1, float* __restrict__ f1max) {
  __shared__ float s[256];
  const float* p = f1 + (size_t)blockIdx.x * 512;
  float m = fmaxf(p[threadIdx.x], p[threadIdx.x + 256]);
  s[threadIdx.x] = m; __syncthreads();
  for (int st = 128; st > 0; st >>= 1) {
    if (threadIdx.x < st) s[threadIdx.x] = fmaxf(s[threadIdx.x], s[threadIdx.x + st]);
    __syncthreads();
  }
  if (threadIdx.x == 0) f1max[blockIdx.x] = s[0];
}

// ---------------------------------------------------------------------------
// K3b: per column j: cmax[j] = LR(f1max + f2[j]); rden[j] = 1/sum_i exp(...)
// ---------------------------------------------------------------------------
__global__ __launch_bounds__(256) void k_colstats(const float* __restrict__ f1, const float* __restrict__ f2,
                                                  const float* __restrict__ f1max,
                                                  float* __restrict__ cmax, float* __restrict__ rden) {
  __shared__ float sf1[512];
  int hb = blockIdx.x;
  const float* F1 = f1 + (size_t)hb * 512;
  const float* F2 = f2 + (size_t)hb * 512;
  sf1[threadIdx.x] = F1[threadIdx.x];
  sf1[threadIdx.x + 256] = F1[threadIdx.x + 256];
  __syncthreads();
  float fm = f1max[hb];
#pragma unroll
  for (int jj = 0; jj < 2; ++jj) {
    int j = threadIdx.x + jj * 256;
    float fj = F2[j];
    float cm = lrelu(fm + fj);
    float s = 0.f;
    for (int i = 0; i < 512; ++i) s += __expf(lrelu(sf1[i] + fj) - cm);
    cmax[(size_t)hb * 512 + j] = cm;
    rden[(size_t)hb * 512 + j] = 1.0f / s;
  }
}

// ---------------------------------------------------------------------------
// K4 (heavy): out1 = softmax-attn @ Wh, elu, scatter into x2[b][i][h*64+d].
// Each wave owns a 16x64 stripe: per K-step one on-the-fly P fragment
// (16 exps/lane -- computed exactly once per (i,j) across the grid) feeds
// 4 independent WMMAs. 1024 blocks = 8h * 32b * 4 rowgroups; 8 waves/group.
// ---------------------------------------------------------------------------
__global__ __launch_bounds__(256) void k_attn1(const unsigned short* __restrict__ whA,
                                               const float* __restrict__ f1, const float* __restrict__ f2,
                                               const float* __restrict__ cmax, const float* __restrict__ rden,
                                               unsigned short* __restrict__ x2) {
  __shared__ float s_f2[512], s_cm[512], s_rd[512];
  int blk = blockIdx.x;
  int h   = blk >> 7;             // / 128
  int rem = blk & 127;
  int b   = rem >> 2;
  int grp = rem & 3;
  int hb  = h * 32 + b;
  for (int t = threadIdx.x; t < 512; t += 256) {
    s_f2[t] = f2[(size_t)hb * 512 + t];
    s_cm[t] = cmax[(size_t)hb * 512 + t];
    s_rd[t] = rden[(size_t)hb * 512 + t];
  }
  __syncthreads();
  int wave = threadIdx.x >> 5, lane = threadIdx.x & 31;
  int mt  = grp * 8 + wave;       // 0..31 within (h,b)
  int i_a = mt * 16 + (lane & 15);
  float f1v = f1[(size_t)hb * 512 + i_a];
  const unsigned short* Wh = whA + ((size_t)h * 16384 + (size_t)b * 512) * 64;
  const int half8 = (lane >> 4) * 8;
  const int kbl   = (lane & 15) + ((lane >> 4) << 4);
  v8f c0 = {}, c1 = {}, c2 = {}, c3 = {};
  for (int kk = 0; kk < 512; kk += 32) {
    v8u au;
#pragma unroll
    for (int v = 0; v < 8; ++v) {
      int k0 = kk + ((v < 4) ? 2 * v : 16 + 2 * (v - 4)) + half8;
      float p0 = __expf(lrelu(f1v + s_f2[k0])     - s_cm[k0])     * s_rd[k0];
      float p1 = __expf(lrelu(f1v + s_f2[k0 + 1]) - s_cm[k0 + 1]) * s_rd[k0 + 1];
      au[v] = (unsigned int)f2bf(p0) | ((unsigned int)f2bf(p1) << 16);
    }
    v8u bu0, bu1, bu2, bu3;
#pragma unroll
    for (int v = 0; v < 8; ++v) {
      int kb = kk + ((v < 4) ? 2 * v : 16 + 2 * (v - 4)) + half8;
      const unsigned short* r0 = Wh + (size_t)kb * 64 + (lane & 15);
      const unsigned short* r1 = r0 + 64;
      bu0[v] = (unsigned int)r0[0]  | ((unsigned int)r1[0]  << 16);
      bu1[v] = (unsigned int)r0[16] | ((unsigned int)r1[16] << 16);
      bu2[v] = (unsigned int)r0[32] | ((unsigned int)r1[32] << 16);
      bu3[v] = (unsigned int)r0[48] | ((unsigned int)r1[48] << 16);
    }
    c0 = wmma_bf16(au, bu0, c0);
    c1 = wmma_bf16(au, bu1, c1);
    c2 = wmma_bf16(au, bu2, c2);
    c3 = wmma_bf16(au, bu3, c3);
  }
  // elu then scatter: x2[b][i][h*64 + d]
  int n = lane & 15;
  int mbase = mt * 16 + ((lane >> 4) << 3);
#pragma unroll
  for (int r = 0; r < 8; ++r) {
    unsigned short* row = x2 + ((size_t)b * 512 + mbase + r) * 512 + h * 64 + n;
    row[0]  = f2bf(eluf(c0[r]));
    row[16] = f2bf(eluf(c1[r]));
    row[32] = f2bf(eluf(c2[r]));
    row[48] = f2bf(eluf(c3[r]));
  }
}

// ---------------------------------------------------------------------------
// K5: Wh2 = x2 @ W_out : plain (16384x512)@(512x64) bf16 WMMA GEMM
// 1024 waves -> 128 blocks.
// ---------------------------------------------------------------------------
__global__ __launch_bounds__(256) void k_gemm2(const unsigned short* __restrict__ x2,
                                               const unsigned short* __restrict__ woutb,
                                               unsigned short* __restrict__ wh2) {
  int wave = (blockIdx.x * blockDim.x + threadIdx.x) >> 5;
  int lane = threadIdx.x & 31;
  int mt = wave;                  // 0..1023
  v8f c0 = {}, c1 = {}, c2 = {}, c3 = {};
  gemm_stripe_n64(x2, 512, woutb, mt * 16, 512, lane, c0, c1, c2, c3);
  store_stripe_bf16(wh2, 64, mt * 16, 0, lane, c0, c1, c2, c3);
}

// ---------------------------------------------------------------------------
// K6: layer-2 f1/f2 dots with a_out
// ---------------------------------------------------------------------------
__global__ void k_fdots2(const unsigned short* __restrict__ wh2, const float* __restrict__ a_out,
                         float* __restrict__ f1b, float* __restrict__ f2b) {
  int idx = blockIdx.x * blockDim.x + threadIdx.x;  // 16384
  const unsigned short* w = wh2 + (size_t)idx * 64;
  float s1 = 0.f, s2 = 0.f;
#pragma unroll 8
  for (int d = 0; d < 64; ++d) { float x = bf2f(w[d]); s1 += x * a_out[d]; s2 += x * a_out[64 + d]; }
  f1b[idx] = s1; f2b[idx] = s2;
}

// ---------------------------------------------------------------------------
// K7: layer-2 attention, only row i=0 is consumed (gat_feats = x[:,0]).
// One block per batch: max-reduce f1, column stats, row-0 weights, 64-d dot.
// ---------------------------------------------------------------------------
__global__ __launch_bounds__(256) void k_attn2(const unsigned short* __restrict__ wh2,
                                               const float* __restrict__ f1b, const float* __restrict__ f2b,
                                               float* __restrict__ gfeat) {
  __shared__ float sf1[512];
  __shared__ float sw[512];
  __shared__ float sred[256];
  int b = blockIdx.x;
  const float* F1 = f1b + (size_t)b * 512;
  const float* F2 = f2b + (size_t)b * 512;
  sf1[threadIdx.x] = F1[threadIdx.x];
  sf1[threadIdx.x + 256] = F1[threadIdx.x + 256];
  __syncthreads();
  sred[threadIdx.x] = fmaxf(sf1[threadIdx.x], sf1[threadIdx.x + 256]);
  __syncthreads();
  for (int st = 128; st > 0; st >>= 1) {
    if (threadIdx.x < st) sred[threadIdx.x] = fmaxf(sred[threadIdx.x], sred[threadIdx.x + st]);
    __syncthreads();
  }
  float fm  = sred[0];
  float f10 = sf1[0];
#pragma unroll
  for (int jj = 0; jj < 2; ++jj) {
    int j = threadIdx.x + jj * 256;
    float fj = F2[j];
    float cm = lrelu(fm + fj);
    float s = 0.f;
    for (int i = 0; i < 512; ++i) s += __expf(lrelu(sf1[i] + fj) - cm);
    sw[j] = __expf(lrelu(f10 + fj) - cm) / s;
  }
  __syncthreads();
  if (threadIdx.x < 64) {
    int d = threadIdx.x;
    float acc = 0.f;
    for (int j = 0; j < 512; ++j) acc += sw[j] * bf2f(wh2[((size_t)b * 512 + j) * 64 + d]);
    gfeat[b * 64 + d] = eluf(acc);
  }
}

// ---------------------------------------------------------------------------
// K8: final MLP: concat -> W1+b1 -> PReLU -> BN -> W2+b2  (tiny, one block)
// ---------------------------------------------------------------------------
__global__ __launch_bounds__(256) void k_mlp(const float* __restrict__ feats, const float* __restrict__ gfeat,
                                             const float* __restrict__ W1, const float* __restrict__ b1,
                                             const float* __restrict__ prelu_a,
                                             const float* __restrict__ gamma, const float* __restrict__ beta,
                                             const float* __restrict__ mean, const float* __restrict__ var,
                                             const float* __restrict__ W2, const float* __restrict__ b2,
                                             float* __restrict__ out) {
  __shared__ float sh[32 * 64];
  for (int t = threadIdx.x; t < 2048; t += 256) {
    int row = t >> 6, d = t & 63;
    float acc = b1[d];
    const float* fr = feats + row * 256;
    for (int f = 0; f < 256; ++f) acc += fr[f] * W1[f * 64 + d];
    const float* gr = gfeat + row * 64;
    for (int f = 0; f < 64; ++f) acc += gr[f] * W1[(256 + f) * 64 + d];
    float a = prelu_a[0];
    acc = acc >= 0.f ? acc : a * acc;
    acc = (acc - mean[d]) * rsqrtf(var[d] + BN_EPS) * gamma[d] + beta[d];
    sh[t] = acc;
  }
  __syncthreads();
  if (threadIdx.x < 32) {
    float acc = b2[0];
    for (int d = 0; d < 64; ++d) acc += sh[threadIdx.x * 64 + d] * W2[d];
    out[threadIdx.x] = acc;
  }
}

// ---------------------------------------------------------------------------
extern "C" void kernel_launch(void* const* d_in, const int* in_sizes, int n_in,
                              void* d_out, int out_size, void* d_ws, size_t ws_size,
                              hipStream_t stream) {
  (void)in_sizes; (void)n_in; (void)out_size; (void)ws_size;
  const float* feats  = (const float*)d_in[0];
  const float* nf     = (const float*)d_in[1];
  const float* wheads = (const float*)d_in[2];
  const float* aheads = (const float*)d_in[3];
  const float* wout   = (const float*)d_in[4];
  const float* aout   = (const float*)d_in[5];
  const float* W1     = (const float*)d_in[6];
  const float* b1     = (const float*)d_in[7];
  const float* prelua = (const float*)d_in[8];
  const float* gamma  = (const float*)d_in[9];
  const float* beta   = (const float*)d_in[10];
  const float* mean   = (const float*)d_in[11];
  const float* var    = (const float*)d_in[12];
  const float* W2     = (const float*)d_in[13];
  const float* b2     = (const float*)d_in[14];

  char* ws = (char*)d_ws;
  size_t off = 0;
  auto alloc = [&](size_t bytes) -> char* {
    char* p = ws + off;
    off = (off + bytes + 255) & ~(size_t)255;
    return p;
  };
  unsigned short* nfb   = (unsigned short*)alloc(4194304ull * 2);  // bf16 neighbor feats
  unsigned short* whhb  = (unsigned short*)alloc(131072ull * 2);   // bf16 W_heads
  unsigned short* woutb = (unsigned short*)alloc(32768ull * 2);    // bf16 W_out
  unsigned short* whA   = (unsigned short*)alloc(8388608ull * 2);  // bf16 Wh (8x16384x64)
  float* f1    = (float*)alloc(131072ull * 4);
  float* f2    = (float*)alloc(131072ull * 4);
  float* f1m   = (float*)alloc(256ull * 4);
  float* cmax  = (float*)alloc(131072ull * 4);
  float* rden  = (float*)alloc(131072ull * 4);
  unsigned short* x2  = (unsigned short*)alloc(8388608ull * 2);    // bf16 (32x512x512)
  unsigned short* wh2 = (unsigned short*)alloc(1048576ull * 2);    // bf16 (16384x64)
  float* f1b   = (float*)alloc(16384ull * 4);
  float* f2b   = (float*)alloc(16384ull * 4);
  float* gfeat = (float*)alloc(2048ull * 4);

  k_convert<<<4194304 / 256, 256, 0, stream>>>(nf, nfb, 4194304);
  k_convert<<<131072 / 256, 256, 0, stream>>>(wheads, whhb, 131072);
  k_convert<<<32768 / 256, 256, 0, stream>>>(wout, woutb, 32768);

  k_gemm1<<<1024, 256, 0, stream>>>(nfb, whhb, whA);              // WMMA, 16x64/wave
  k_fdots<<<512, 256, 0, stream>>>(whA, aheads, f1, f2);
  k_f1max<<<256, 256, 0, stream>>>(f1, f1m);
  k_colstats<<<256, 256, 0, stream>>>(f1, f2, f1m, cmax, rden);
  k_attn1<<<1024, 256, 0, stream>>>(whA, f1, f2, cmax, rden, x2); // WMMA (heavy), 16x64/wave
  k_gemm2<<<128, 256, 0, stream>>>(x2, woutb, wh2);               // WMMA, 16x64/wave
  k_fdots2<<<64, 256, 0, stream>>>(wh2, aout, f1b, f2b);
  k_attn2<<<32, 256, 0, stream>>>(wh2, f1b, f2b, gfeat);
  k_mlp<<<1, 256, 0, stream>>>(feats, gfeat, W1, b1, prelua, gamma, beta, mean, var, W2, b2,
                               (float*)d_out);
}
